// MLHAttention_65283502899669
// MI455X (gfx1250) — matrix operand order repaired and checked by reading, not compiled, not run on hardware
//
#include <hip/hip_runtime.h>

// ---------------- problem constants (B=1) ----------------
#define S_LEN 4096
#define HID   1024
#define NHEAD 16
#define NKVH  4
#define HDIM  64
#define KVW   (NKVH * HDIM)   // 256

typedef __attribute__((ext_vector_type(16))) __bf16          bf16x16;
typedef __attribute__((ext_vector_type(8)))  float           f32x8;
typedef __attribute__((ext_vector_type(8)))  unsigned short  ushort8;
typedef __attribute__((ext_vector_type(16))) unsigned short  ushort16;
typedef __attribute__((ext_vector_type(4)))  int             v4i;

#if __has_builtin(__builtin_amdgcn_global_load_async_to_lds_b128) && \
    __has_builtin(__builtin_amdgcn_s_wait_asynccnt)
#define ASYNC_LDS 1
#else
#define ASYNC_LDS 0
#endif

static __device__ __forceinline__ unsigned short f32_to_bf16_rne(float f) {
  unsigned int u = __builtin_bit_cast(unsigned int, f);
  u += 0x7FFFu + ((u >> 16) & 1u);
  return (unsigned short)(u >> 16);
}

// Build a 16-element bf16 WMMA fragment from two 16-byte chunks.
static __device__ __forceinline__ bf16x16 frag16(const unsigned short* p0,
                                                 const unsigned short* p1) {
  ushort8 lo = *(const ushort8*)p0;
  ushort8 hi = *(const ushort8*)p1;
  ushort16 u = __builtin_shufflevector(lo, hi, 0,1,2,3,4,5,6,7,8,9,10,11,12,13,14,15);
  return __builtin_bit_cast(bf16x16, u);
}

static __device__ __forceinline__ f32x8 zero8() {
  f32x8 v = {0.f,0.f,0.f,0.f,0.f,0.f,0.f,0.f};
  return v;
}

// 16-byte global -> LDS copy: async (gfx1250 GLOBAL_LOAD_ASYNC_TO_LDS_B128,
// tracked by ASYNCcnt) when available, else synchronous load + ds_store.
// Builtin proto (from hipcc diagnostic): (v4i AS1* src, v4i AS3* dst, Ii, Ii).
static __device__ __forceinline__ void stage16(unsigned short* lds_dst,
                                               const unsigned short* gsrc) {
#if ASYNC_LDS
  __builtin_amdgcn_global_load_async_to_lds_b128(
      (__attribute__((address_space(1))) v4i*)gsrc,
      (__attribute__((address_space(3))) v4i*)lds_dst, 0, 0);
#else
  *(ushort8*)lds_dst = *(const ushort8*)gsrc;
#endif
}

static __device__ __forceinline__ void wait_stage() {
#if ASYNC_LDS
  __builtin_amdgcn_s_wait_asynccnt(0);
#endif
}

// ---------------- f32 -> bf16 cast ----------------
__global__ void cast_f32_to_bf16(const float* __restrict__ in,
                                 unsigned short* __restrict__ out, int n) {
  int i = blockIdx.x * blockDim.x + threadIdx.x;
  int stride = gridDim.x * blockDim.x;
  for (; i < n; i += stride) out[i] = f32_to_bf16_rne(in[i]);
}

// ---------------- C[M,N] = A[M,K] @ B[N,K]^T  (bf16 in, f32 acc) ----------------
// block = 256 threads = 8 waves; wave -> 16 rows x 64 cols; grid = (N/64, M/128).
// B tile (64n x 64k, shared by all 8 waves) is staged in LDS, double buffered,
// via async global->LDS copies. A fragments are per-wave unique -> direct loads.
template<bool BF16OUT>
__global__ __launch_bounds__(256) void gemm_a_x_bT(
    const unsigned short* __restrict__ A, const unsigned short* __restrict__ B,
    void* __restrict__ C, int K, int lda, int ldb, int ldc) {
  __shared__ __align__(16) unsigned short Bt[2][64 * 64];  // 8KB per buffer

  const int t    = threadIdx.x;
  const int lane = t & 31;
  const int wv   = t >> 5;
  const int hl   = lane >> 4;    // wave half (0/1)
  const int lr   = lane & 15;
  const int m0 = blockIdx.y * 128 + wv * 16;
  const int n0 = blockIdx.x * 64;

  f32x8 acc[4] = { zero8(), zero8(), zero8(), zero8() };
  const unsigned short* arow = A + (size_t)(m0 + lr) * lda;

  // Stage helper: copy B rows n0..n0+63, cols ks..ks+63 into Bt[buf].
  // 512 16B chunks; thread t copies chunks t and t+256. Chunk c: row c>>3,
  // in-row 16B slot c&7; LDS layout row-major [64][64] -> byte offset c*16.
  auto stageB = [&](int buf, int ks) {
#pragma unroll
    for (int i = 0; i < 2; ++i) {
      int c = t + i * 256;
      stage16(&Bt[buf][c * 8],
              B + (size_t)(n0 + (c >> 3)) * ldb + ks + (c & 7) * 8);
    }
  };

  stageB(0, 0);
  int buf = 0;
  for (int ks = 0; ks < K; ks += 64, buf ^= 1) {
    wait_stage();
    __syncthreads();
    if (ks + 64 < K) stageB(buf ^ 1, ks + 64);
    if (ks + 64 < K) __builtin_prefetch(arow + ks + 64, 0, 1);  // global_prefetch_b8
#pragma unroll
    for (int kc = 0; kc < 2; ++kc) {
      int k = ks + kc * 32;
      // A fragment: row (m0+lr), K chunks {16*hl..}, {16*hl+8..}
      bf16x16 a = frag16(arow + k + 16 * hl, arow + k + 16 * hl + 8);
#pragma unroll
      for (int nt = 0; nt < 4; ++nt) {
        const unsigned short* brow = &Bt[buf][(nt * 16 + lr) * 64 + kc * 32 + 16 * hl];
        bf16x16 b = frag16(brow, brow + 8);
        acc[nt] = __builtin_amdgcn_wmma_f32_16x16x32_bf16(
            false, a, false, b, (short)0, acc[nt], false, false);
      }
    }
  }
#pragma unroll
  for (int nt = 0; nt < 4; ++nt)
#pragma unroll
    for (int r = 0; r < 8; ++r) {
      size_t idx = (size_t)(m0 + 8 * hl + r) * ldc + n0 + nt * 16 + lr;
      if (BF16OUT) ((unsigned short*)C)[idx] = f32_to_bf16_rne(acc[nt][r]);
      else         ((float*)C)[idx]          = acc[nt][r];
    }
}

// ---------------- fused flash attention ----------------
// grid = (S/128, NHEAD), block = 256 (8 waves). Each wave: 16 query rows.
// Streams 64 key columns / iteration. K and V^T tiles (shared by all waves)
// are staged into LDS via async copies, double buffered. P tiles are
// wave-private (LDS ops from one wave are in-order -> no block barrier).
__global__ __launch_bounds__(256) void flash_attn(
    const unsigned short* __restrict__ Qb, const unsigned short* __restrict__ Kb,
    const unsigned short* __restrict__ Vtb, const float* __restrict__ mask,
    unsigned short* __restrict__ Ob) {
  __shared__ __align__(16) unsigned short Kt[2][64 * 64];  // [k-row][hd]
  __shared__ __align__(16) unsigned short Vt[2][64 * 64];  // [hd][k-col]
  __shared__ __align__(16) unsigned short Pl[8][16 * 64];  // per-wave P tile

  const int t    = threadIdx.x;
  const int lane = t & 31;
  const int wv   = t >> 5;
  const int hl   = lane >> 4;
  const int lr   = lane & 15;
  const int h    = blockIdx.y;
  const int hkv  = h >> 2;                       // GQA: 4 Q heads per KV head
  const int q0   = blockIdx.x * 128 + wv * 16;   // this wave's query-row base

  // Q A-fragments live in registers for the whole kernel (2 chunks of hd=32)
  bf16x16 aQ[2];
  const unsigned short* qrow = Qb + (size_t)(q0 + lr) * HID + h * HDIM;
#pragma unroll
  for (int c = 0; c < 2; ++c)
    aQ[c] = frag16(qrow + c * 32 + 16 * hl, qrow + c * 32 + 16 * hl + 8);

  f32x8 o[4] = { zero8(), zero8(), zero8(), zero8() };
  float m_run[8], l_run[8];
#pragma unroll
  for (int r = 0; r < 8; ++r) { m_run[r] = -1e30f; l_run[r] = 0.f; }
  const float scale = 0.125f;  // 1/sqrt(64)

  // Stage K rows kb..kb+63 (cols hkv*64..+63) and V^T rows hkv*64..+63
  // (cols kb..kb+63). Each tile: 512 16B chunks, 2 per thread.
  auto stageKV = [&](int buf, int kb) {
#pragma unroll
    for (int i = 0; i < 2; ++i) {
      int c = t + i * 256;
      stage16(&Kt[buf][c * 8],
              Kb + (size_t)(kb + (c >> 3)) * KVW + hkv * HDIM + (c & 7) * 8);
      stage16(&Vt[buf][c * 8],
              Vtb + (size_t)(hkv * HDIM + (c >> 3)) * S_LEN + kb + (c & 7) * 8);
    }
  };

  stageKV(0, 0);
  int buf = 0;
  for (int kb = 0; kb < S_LEN; kb += 64, buf ^= 1) {
    wait_stage();
    __syncthreads();
    if (kb + 64 < S_LEN) stageKV(buf ^ 1, kb + 64);

    // ---- scores: S = Q @ K^T (16x64 tile via 4 n-tiles x 2 hd-chunks) ----
    f32x8 s[4] = { zero8(), zero8(), zero8(), zero8() };
#pragma unroll
    for (int nt = 0; nt < 4; ++nt) {
#pragma unroll
      for (int c = 0; c < 2; ++c) {
        const unsigned short* krow = &Kt[buf][(nt * 16 + lr) * 64 + c * 32 + 16 * hl];
        bf16x16 bK = frag16(krow, krow + 8);
        s[nt] = __builtin_amdgcn_wmma_f32_16x16x32_bf16(
            false, aQ[c], false, bK, (short)0, s[nt], false, false);
      }
    }
    // ---- scale + attention mask ----
#pragma unroll
    for (int nt = 0; nt < 4; ++nt)
#pragma unroll
      for (int r = 0; r < 8; ++r) {
        int qm = q0 + 8 * hl + r;
        float mval = mask[(size_t)qm * S_LEN + kb + nt * 16 + lr];
        s[nt][r] = s[nt][r] * scale + mval;
      }
    // ---- online softmax (row reductions across 16-lane halves) ----
#pragma unroll
    for (int r = 0; r < 8; ++r) {
      float bm = s[0][r];
#pragma unroll
      for (int nt = 1; nt < 4; ++nt) bm = fmaxf(bm, s[nt][r]);
      for (int m = 1; m < 16; m <<= 1) bm = fmaxf(bm, __shfl_xor(bm, m, 32));
      float mn = fmaxf(m_run[r], bm);
      float alpha = __expf(m_run[r] - mn);
      m_run[r] = mn;
      float rs = 0.f;
#pragma unroll
      for (int nt = 0; nt < 4; ++nt) {
        float pv = __expf(s[nt][r] - mn);
        s[nt][r] = pv;
        rs += pv;
      }
      for (int m = 1; m < 16; m <<= 1) rs += __shfl_xor(rs, m, 32);
      l_run[r] = l_run[r] * alpha + rs;
#pragma unroll
      for (int nt = 0; nt < 4; ++nt) o[nt][r] *= alpha;
    }
    // ---- stage P to wave-private LDS tile (bf16, 16x64 row-major).
    // LDS ops from one wave execute in order -> only a compile-time
    // scheduling fence is needed between the stores and the reads.
#pragma unroll
    for (int nt = 0; nt < 4; ++nt)
#pragma unroll
      for (int r = 0; r < 8; ++r)
        Pl[wv][(8 * hl + r) * 64 + nt * 16 + lr] = f32_to_bf16_rne(s[nt][r]);
    __builtin_amdgcn_sched_barrier(0);
    // ---- O += P @ V  (V^T tile in LDS -> clean B-operands) ----
#pragma unroll
    for (int c = 0; c < 2; ++c) {
      const unsigned short* pr = &Pl[wv][lr * 64 + c * 32];
      bf16x16 aP = frag16(pr + 16 * hl, pr + 16 * hl + 8);
#pragma unroll
      for (int nt = 0; nt < 4; ++nt) {
        const unsigned short* vrow = &Vt[buf][(nt * 16 + lr) * 64 + c * 32 + 16 * hl];
        bf16x16 bV = frag16(vrow, vrow + 8);
        o[nt] = __builtin_amdgcn_wmma_f32_16x16x32_bf16(
            false, aP, false, bV, (short)0, o[nt], false, false);
      }
    }
  }
  // ---- normalize, store O as bf16 (A-operand for the output projection) ----
#pragma unroll
  for (int nt = 0; nt < 4; ++nt)
#pragma unroll
    for (int r = 0; r < 8; ++r) {
      float v = o[nt][r] / l_run[r];
      Ob[(size_t)(q0 + 8 * hl + r) * HID + h * HDIM + nt * 16 + lr] =
          f32_to_bf16_rne(v);
    }
}

// ---------------- host-side orchestration ----------------
extern "C" void kernel_launch(void* const* d_in, const int* in_sizes, int n_in,
                              void* d_out, int out_size, void* d_ws, size_t ws_size,
                              hipStream_t stream) {
  (void)in_sizes; (void)n_in; (void)out_size; (void)ws_size;
  const float* X    = (const float*)d_in[0];  // [S, HID]
  const float* mask = (const float*)d_in[1];  // [S, S]
  const float* Wq   = (const float*)d_in[2];  // [HID, HID]
  const float* Wk   = (const float*)d_in[3];  // [KVW, HID]
  const float* Wv   = (const float*)d_in[4];  // [KVW, HID]
  const float* Wo   = (const float*)d_in[5];  // [HID, HID]

  unsigned short* ws = (unsigned short*)d_ws;
  const size_t nX  = (size_t)S_LEN * HID;   // 4M
  const size_t nWq = (size_t)HID * HID;     // 1M
  const size_t nWk = (size_t)KVW * HID;     // 256K
  unsigned short* Xb  = ws;              size_t off = nX;
  unsigned short* Wqb = ws + off;        off += nWq;
  unsigned short* Wkb = ws + off;        off += nWk;
  unsigned short* Wvb = ws + off;        off += nWk;
  unsigned short* Wob = ws + off;        off += nWq;
  unsigned short* Qb  = ws + off;        off += nX;                    // [S, HID]
  unsigned short* Kb  = ws + off;        off += (size_t)S_LEN * KVW;   // [S, KVW]
  unsigned short* Vtb = ws + off;        off += (size_t)KVW * S_LEN;   // [KVW, S]
  unsigned short* Ob  = ws + off;        /* [S, HID] */

  // 1) casts to bf16
  cast_f32_to_bf16<<<4096, 256, 0, stream>>>(X,  Xb,  (int)nX);
  cast_f32_to_bf16<<<2048, 256, 0, stream>>>(Wq, Wqb, (int)nWq);
  cast_f32_to_bf16<<<1024, 256, 0, stream>>>(Wk, Wkb, (int)nWk);
  cast_f32_to_bf16<<<1024, 256, 0, stream>>>(Wv, Wvb, (int)nWk);
  cast_f32_to_bf16<<<2048, 256, 0, stream>>>(Wo, Wob, (int)nWq);

  // 2) projections
  // Qb = Xb @ Wq^T : M=S, N=HID, K=HID
  gemm_a_x_bT<true><<<dim3(HID / 64, S_LEN / 128), 256, 0, stream>>>(
      Xb, Wqb, Qb, HID, HID, HID, HID);
  // Kb = Xb @ Wk^T : M=S, N=KVW, K=HID
  gemm_a_x_bT<true><<<dim3(KVW / 64, S_LEN / 128), 256, 0, stream>>>(
      Xb, Wkb, Kb, HID, HID, HID, KVW);
  // Vtb = Wv @ X^T : M=KVW, N=S, K=HID  (A=Wvb, "weights"=Xb)
  gemm_a_x_bT<true><<<dim3(S_LEN / 64, KVW / 128), 256, 0, stream>>>(
      Wvb, Xb, Vtb, HID, HID, HID, S_LEN);

  // 3) fused attention (flash-style, online softmax)
  flash_attn<<<dim3(S_LEN / 128, NHEAD), 256, 0, stream>>>(Qb, Kb, Vtb, mask, Ob);

  // 4) output projection: out = Ob @ Wo^T (f32 out)
  gemm_a_x_bT<false><<<dim3(HID / 64, S_LEN / 128), 256, 0, stream>>>(
      Ob, Wob, (float*)d_out, HID, HID, HID, HID);
}
